// VectorQuantizer_81595788689815
// MI455X (gfx1250) — compile-verified, matrix operand order checked by reading, not run on hardware
//
#include <hip/hip_runtime.h>
#include <hip/hip_bf16.h>

// ---------------------------------------------------------------------------
// Gumbel-softmax vector quantizer for MI455X (gfx1250), wave32 + bf16 WMMA.
//   N=4096 tokens, D=512, K=8192 codes.
// GEMM1 (logits) and GEMM2 (probs @ codebook) run on v_wmma_f32_16x16x32_bf16
// with fp32 accumulation. Codebook is L2-resident (8 MB bf16 << 192 MB L2).
// The WMMA loops are software-pipelined and the schedule is *pinned* with
// __builtin_amdgcn_sched_group_barrier so fragment loads run 1-2 groups
// ahead of the WMMAs that consume them (the plain-source pipeline was
// re-serialized by the machine scheduler in previous rounds).
// ---------------------------------------------------------------------------

#define N_TOK 4096
#define DIMD  512
#define KCB   8192
#define NEGINF (-3.402823466e38f)

// sched_group_barrier masks
#define SGB_WMMA  0x008
#define SGB_VMRD  0x020

typedef __attribute__((ext_vector_type(16))) __bf16 bf16x16;
typedef __attribute__((ext_vector_type(8)))  float  floatx8;

// 16-bit A/B fragment for 16x16x32 WMMA (wave32):
//   lane L holds row/col (L&15); lanes 0-15 take K in {0..7}U{16..23},
//   lanes 16-31 take K in {8..15}U{24..31}  -> two contiguous 16B chunks.
__device__ __forceinline__ bf16x16 load_frag(const __bf16* base32, int khalf) {
  union { bf16x16 v; uint4 q[2]; } u;
  u.q[0] = *reinterpret_cast<const uint4*>(base32 + khalf);
  u.q[1] = *reinterpret_cast<const uint4*>(base32 + khalf + 16);
  return u.v;
}

__device__ __forceinline__ float gumbelf(float u) {
  // -log(-log(u + eps) + eps)
  return -__logf(-__logf(u + 1e-10f) + 1e-10f);
}

#define WMMA_BF16(a, b, c) \
  __builtin_amdgcn_wmma_f32_16x16x32_bf16(false, (a), false, (b), (short)0, (c), false, false)

// ---------------------------------------------------------------------------
// Prep: x -> bf16, row sums of x^2; also zero the 3 scalar output slots.
// ---------------------------------------------------------------------------
__global__ void __launch_bounds__(128) prep_x_kernel(
    const float* __restrict__ x, __bf16* __restrict__ xb,
    float* __restrict__ xsq, float* __restrict__ outScal) {
  __shared__ float red[128];
  const int row = blockIdx.x;
  const int tid = threadIdx.x;
  if (row == 0 && tid < 3) outScal[tid] = 0.f;   // loss, loss, entropy accumulators
  const float* xr = x + (size_t)row * DIMD;
  __bf16* xbr = xb + (size_t)row * DIMD;
  float s = 0.f;
  for (int d = tid; d < DIMD; d += 128) {
    const float v = xr[d];
    s += v * v;
    xbr[d] = (__bf16)v;
  }
  red[tid] = s;
  __syncthreads();
  for (int k = 64; k > 0; k >>= 1) { if (tid < k) red[tid] += red[tid + k]; __syncthreads(); }
  if (tid == 0) xsq[row] = red[0];
}

// ---------------------------------------------------------------------------
// Prep: codebook -> bf16 (row-major + transposed cbT[d][k]), row sums of c^2.
// The transpose makes GEMM2 B-fragments contiguous 16B loads per lane.
// ---------------------------------------------------------------------------
__global__ void __launch_bounds__(128) prep_cb_kernel(
    const float* __restrict__ cb, __bf16* __restrict__ cbb,
    __bf16* __restrict__ cbT, float* __restrict__ csq) {
  __shared__ float red[128];
  const int row = blockIdx.x;          // codebook entry index k
  const int tid = threadIdx.x;
  const float* cr = cb + (size_t)row * DIMD;
  __bf16* cbr = cbb + (size_t)row * DIMD;
  float s = 0.f;
  for (int d = tid; d < DIMD; d += 128) {
    const float v = cr[d];
    s += v * v;
    const __bf16 bv = (__bf16)v;
    cbr[d] = bv;
    cbT[(size_t)d * KCB + row] = bv;
  }
  red[tid] = s;
  __syncthreads();
  for (int k = 64; k > 0; k >>= 1) { if (tid < k) red[tid] += red[tid + k]; __syncthreads(); }
  if (tid == 0) csq[row] = red[0];
}

// Pipelined 16-chunk dual-tile contraction over D=512.
// Schedule pinned to: [ld4][ld4] ([wmma2][ld4])x14 [wmma2][wmma2]
// -> every WMMA pair waits only on loads issued two chunk-groups earlier.
__device__ __forceinline__ void gemm1_tilepair(
    const bf16x16* A, const __bf16* crow0, const __bf16* crow1, int khalf,
    floatx8& acc0, floatx8& acc1) {
  __builtin_amdgcn_sched_barrier(0);
  bf16x16 b0 = load_frag(crow0, khalf);
  bf16x16 b1 = load_frag(crow1, khalf);
#pragma unroll
  for (int c = 0; c < 15; ++c) {
    const bf16x16 n0 = load_frag(crow0 + (c + 1) * 32, khalf);
    const bf16x16 n1 = load_frag(crow1 + (c + 1) * 32, khalf);
    acc0 = WMMA_BF16(A[c], b0, acc0);
    acc1 = WMMA_BF16(A[c], b1, acc1);
    b0 = n0;
    b1 = n1;
  }
  acc0 = WMMA_BF16(A[15], b0, acc0);
  acc1 = WMMA_BF16(A[15], b1, acc1);
  // Pin the pipeline: 2-chunk load lookahead ahead of each WMMA pair.
  __builtin_amdgcn_sched_group_barrier(SGB_VMRD, 4, 0);   // loads chunk 0
  __builtin_amdgcn_sched_group_barrier(SGB_VMRD, 4, 0);   // loads chunk 1
#pragma unroll
  for (int c = 0; c < 14; ++c) {
    __builtin_amdgcn_sched_group_barrier(SGB_WMMA, 2, 0); // WMMAs chunk c
    __builtin_amdgcn_sched_group_barrier(SGB_VMRD, 4, 0); // loads chunk c+2
  }
  __builtin_amdgcn_sched_group_barrier(SGB_WMMA, 2, 0);   // WMMAs chunk 14
  __builtin_amdgcn_sched_group_barrier(SGB_WMMA, 2, 0);   // WMMAs chunk 15
  __builtin_amdgcn_sched_barrier(0);
}

// ---------------------------------------------------------------------------
// Fused logits + gumbel-softmax stats. One workgroup owns 16 token rows and
// all K=8192 codes. 4 waves; each wave works two adjacent 16x16 column tiles
// per macro-step (two independent, software-pipelined load/WMMA chains).
// Pass 1: row-max of z1=(logits+g1)/tau and argmax of logits+g2 (indices).
// Pass 2: recompute logits (codebook is L2-resident), emit pe=exp(z1-max)
//         as bf16, accumulate S=sum pe and W=sum pe*(z1-max).
// entropy_row = log S - W/S ;  invS stored for GEMM2 normalization.
// ---------------------------------------------------------------------------
__global__ void __launch_bounds__(128) vq_logits_softmax_kernel(
    const float* __restrict__ u1, const float* __restrict__ u2,
    const __bf16* __restrict__ xb, const __bf16* __restrict__ cbb,
    const float* __restrict__ xsq, const float* __restrict__ csq,
    __bf16* __restrict__ P, float* __restrict__ invS,
    float* __restrict__ outIdx, float* __restrict__ outEnt) {
  __shared__ float s_a[1024];
  __shared__ float s_b[1024];
  __shared__ int   s_i[1024];
  __shared__ float rowMax[16];

  const int row0  = blockIdx.x * 16;
  const int tid   = threadIdx.x;
  const int lane  = tid & 31;
  const int wave  = tid >> 5;
  const int m     = lane & 15;
  const int hi    = lane >> 4;
  const int khalf = hi * 8;

  // Hoist the 16x512 bf16 A-operand (x rows) into registers: 16 k-chunks.
  bf16x16 A[16];
  {
    const __bf16* xrow = xb + (size_t)(row0 + m) * DIMD;
#pragma unroll
    for (int c = 0; c < 16; ++c) A[c] = load_frag(xrow + c * 32, khalf);
  }
  float xsqv[8];
#pragma unroll
  for (int r = 0; r < 8; ++r) xsqv[r] = xsq[row0 + r + 8 * hi];

  float zmax[8], ymax[8];
  int   yidx[8];
#pragma unroll
  for (int r = 0; r < 8; ++r) { zmax[r] = NEGINF; ymax[r] = NEGINF; yidx[r] = 0; }

  // ---- Pass 1: maxima / argmax --------------------------------------------
  // Each wave: 32 columns per macro-step (two tiles), 64 macro-steps.
  for (int t = 0; t < KCB / 128; ++t) {
    const int n0   = t * 128 + wave * 32;      // first tile base
    const int col0 = n0 + m;
    const int col1 = n0 + 16 + m;
    floatx8 acc0 = {0.f, 0.f, 0.f, 0.f, 0.f, 0.f, 0.f, 0.f};
    floatx8 acc1 = {0.f, 0.f, 0.f, 0.f, 0.f, 0.f, 0.f, 0.f};
    gemm1_tilepair(A, cbb + (size_t)col0 * DIMD, cbb + (size_t)col1 * DIMD,
                   khalf, acc0, acc1);
    const float cs0 = csq[col0];
    const float cs1 = csq[col1];
#pragma unroll
    for (int r = 0; r < 8; ++r) {
      const int row = row0 + r + 8 * hi;
      const size_t off0 = (size_t)row * KCB + col0;
      const size_t off1 = (size_t)row * KCB + col1;
      // prefetch next macro-step's noise lines (streaming 128MB buffers)
      __builtin_prefetch(&u1[off0 + 128], 0, 1);
      __builtin_prefetch(&u2[off0 + 128], 0, 1);
      {
        const float logit = 2.f * acc0[r] - xsqv[r] - cs0;
        const float z = (logit + gumbelf(u1[off0])) * 0.5f;   // / TAU
        const float y = logit + gumbelf(u2[off0]);
        zmax[r] = fmaxf(zmax[r], z);
        if (y > ymax[r] || (y == ymax[r] && col0 < yidx[r])) { ymax[r] = y; yidx[r] = col0; }
      }
      {
        const float logit = 2.f * acc1[r] - xsqv[r] - cs1;
        const float z = (logit + gumbelf(u1[off1])) * 0.5f;
        const float y = logit + gumbelf(u2[off1]);
        zmax[r] = fmaxf(zmax[r], z);
        if (y > ymax[r] || (y == ymax[r] && col1 < yidx[r])) { ymax[r] = y; yidx[r] = col1; }
      }
    }
  }

  // Cross-lane / cross-wave reduction of pass-1 stats via LDS.
#pragma unroll
  for (int r = 0; r < 8; ++r) {
    const int base = (wave * 32 + lane) * 8 + r;
    s_a[base] = zmax[r]; s_b[base] = ymax[r]; s_i[base] = yidx[r];
  }
  __syncthreads();
  if (tid < 16) {                       // thread t reduces token row t
    const int rhi = tid >> 3, rr = tid & 7;
    float zm = NEGINF, ym = NEGINF; int yi = 0;
    for (int w = 0; w < 4; ++w)
      for (int l = 0; l < 16; ++l) {
        const int idx = (w * 32 + rhi * 16 + l) * 8 + rr;
        zm = fmaxf(zm, s_a[idx]);
        const float yv = s_b[idx]; const int ii = s_i[idx];
        if (yv > ym || (yv == ym && ii < yi)) { ym = yv; yi = ii; }
      }
    rowMax[tid] = zm;
    outIdx[row0 + tid] = (float)yi;
  }
  __syncthreads();

  float zm_r[8];
#pragma unroll
  for (int r = 0; r < 8; ++r) zm_r[r] = rowMax[r + 8 * hi];

  // ---- Pass 2: exp / store pe / partial S,W -------------------------------
  float Sacc[8], Wacc[8];
#pragma unroll
  for (int r = 0; r < 8; ++r) { Sacc[r] = 0.f; Wacc[r] = 0.f; }

  for (int t = 0; t < KCB / 128; ++t) {
    const int n0   = t * 128 + wave * 32;
    const int col0 = n0 + m;
    const int col1 = n0 + 16 + m;
    floatx8 acc0 = {0.f, 0.f, 0.f, 0.f, 0.f, 0.f, 0.f, 0.f};
    floatx8 acc1 = {0.f, 0.f, 0.f, 0.f, 0.f, 0.f, 0.f, 0.f};
    gemm1_tilepair(A, cbb + (size_t)col0 * DIMD, cbb + (size_t)col1 * DIMD,
                   khalf, acc0, acc1);
    const float cs0 = csq[col0];
    const float cs1 = csq[col1];
#pragma unroll
    for (int r = 0; r < 8; ++r) {
      const int row = row0 + r + 8 * hi;
      const size_t off0 = (size_t)row * KCB + col0;
      const size_t off1 = (size_t)row * KCB + col1;
      __builtin_prefetch(&u1[off0 + 128], 0, 1);
      {
        const float logit = 2.f * acc0[r] - xsqv[r] - cs0;
        const float z  = (logit + gumbelf(u1[off0])) * 0.5f;
        const float tt = z - zm_r[r];
        const float pe = __expf(tt);
        Sacc[r] += pe;
        Wacc[r] += pe * tt;
        P[off0] = (__bf16)pe;
      }
      {
        const float logit = 2.f * acc1[r] - xsqv[r] - cs1;
        const float z  = (logit + gumbelf(u1[off1])) * 0.5f;
        const float tt = z - zm_r[r];
        const float pe = __expf(tt);
        Sacc[r] += pe;
        Wacc[r] += pe * tt;
        P[off1] = (__bf16)pe;
      }
    }
  }

#pragma unroll
  for (int r = 0; r < 8; ++r) {
    const int base = (wave * 32 + lane) * 8 + r;
    s_a[base] = Sacc[r]; s_b[base] = Wacc[r];
  }
  __syncthreads();
  if (tid < 16) {
    const int rhi = tid >> 3, rr = tid & 7;
    float S = 0.f, W = 0.f;
    for (int w = 0; w < 4; ++w)
      for (int l = 0; l < 16; ++l) {
        const int idx = (w * 32 + rhi * 16 + l) * 8 + rr;
        S += s_a[idx]; W += s_b[idx];
      }
    invS[row0 + tid] = 1.f / S;
    const float ent = __logf(S) - W / S;       // -sum p log p
    atomicAdd(outEnt, ent * (1.0f / (float)N_TOK));
  }
}

// ---------------------------------------------------------------------------
// GEMM2: quantized = (pe @ cb) * invS, fused MSE-loss reduction.
// Workgroup: 16 rows x 512 cols, 8 waves x 4 tiles, contraction K=8192.
// Unroll-by-2 software pipeline, pinned to [ld10][wmma4][ld10][wmma4] per
// iteration so each WMMA group waits only on loads issued a group earlier.
// ---------------------------------------------------------------------------
__global__ void __launch_bounds__(256) vq_gemm2_kernel(
    const __bf16* __restrict__ P, const __bf16* __restrict__ cbT,
    const float* __restrict__ invS, const float* __restrict__ x,
    float* __restrict__ outQ, float* __restrict__ outLoss) {
  __shared__ float red[256];
  const int row0  = blockIdx.x * 16;
  const int tid   = threadIdx.x;
  const int lane  = tid & 31;
  const int wave  = tid >> 5;
  const int m     = lane & 15;
  const int hi    = lane >> 4;
  const int khalf = hi * 8;
  const int dbase = wave * 64;

  const floatx8 zero8 = {0.f, 0.f, 0.f, 0.f, 0.f, 0.f, 0.f, 0.f};
  floatx8 acc[4] = {zero8, zero8, zero8, zero8};

  const __bf16* prow  = P   + (size_t)(row0 + m) * KCB;
  const __bf16* brow0 = cbT + (size_t)(dbase +  0 + m) * KCB;
  const __bf16* brow1 = cbT + (size_t)(dbase + 16 + m) * KCB;
  const __bf16* brow2 = cbT + (size_t)(dbase + 32 + m) * KCB;
  const __bf16* brow3 = cbT + (size_t)(dbase + 48 + m) * KCB;

  const int NSTEP = KCB / 32;   // 256 (even)
  // stage X preloaded for step 0
  bf16x16 ax  = load_frag(prow,  khalf);
  bf16x16 x0  = load_frag(brow0, khalf);
  bf16x16 x1  = load_frag(brow1, khalf);
  bf16x16 x2  = load_frag(brow2, khalf);
  bf16x16 x3  = load_frag(brow3, khalf);
  for (int s = 0; s < NSTEP - 2; s += 2) {
    const int k1 = (s + 1) * 32;
    // issue stage-Y loads before retiring stage-X WMMAs
    bf16x16 ay  = load_frag(prow  + k1, khalf);
    bf16x16 y0  = load_frag(brow0 + k1, khalf);
    bf16x16 y1  = load_frag(brow1 + k1, khalf);
    bf16x16 y2  = load_frag(brow2 + k1, khalf);
    bf16x16 y3  = load_frag(brow3 + k1, khalf);
    acc[0] = WMMA_BF16(ax, x0, acc[0]);
    acc[1] = WMMA_BF16(ax, x1, acc[1]);
    acc[2] = WMMA_BF16(ax, x2, acc[2]);
    acc[3] = WMMA_BF16(ax, x3, acc[3]);
    const int k2 = (s + 2) * 32;
    ax = load_frag(prow  + k2, khalf);
    x0 = load_frag(brow0 + k2, khalf);
    x1 = load_frag(brow1 + k2, khalf);
    x2 = load_frag(brow2 + k2, khalf);
    x3 = load_frag(brow3 + k2, khalf);
    acc[0] = WMMA_BF16(ay, y0, acc[0]);
    acc[1] = WMMA_BF16(ay, y1, acc[1]);
    acc[2] = WMMA_BF16(ay, y2, acc[2]);
    acc[3] = WMMA_BF16(ay, y3, acc[3]);
    // Pin the pipeline inside the loop body.
    __builtin_amdgcn_sched_group_barrier(SGB_VMRD, 10, 0); // Y loads
    __builtin_amdgcn_sched_group_barrier(SGB_WMMA,  4, 0); // X WMMAs
    __builtin_amdgcn_sched_group_barrier(SGB_VMRD, 10, 0); // next X loads
    __builtin_amdgcn_sched_group_barrier(SGB_WMMA,  4, 0); // Y WMMAs
  }
  {
    // tail: steps NSTEP-2 (in X) and NSTEP-1
    const int k1 = (NSTEP - 1) * 32;
    const bf16x16 ay = load_frag(prow  + k1, khalf);
    const bf16x16 y0 = load_frag(brow0 + k1, khalf);
    const bf16x16 y1 = load_frag(brow1 + k1, khalf);
    const bf16x16 y2 = load_frag(brow2 + k1, khalf);
    const bf16x16 y3 = load_frag(brow3 + k1, khalf);
    acc[0] = WMMA_BF16(ax, x0, acc[0]);
    acc[1] = WMMA_BF16(ax, x1, acc[1]);
    acc[2] = WMMA_BF16(ax, x2, acc[2]);
    acc[3] = WMMA_BF16(ax, x3, acc[3]);
    acc[0] = WMMA_BF16(ay, y0, acc[0]);
    acc[1] = WMMA_BF16(ay, y1, acc[1]);
    acc[2] = WMMA_BF16(ay, y2, acc[2]);
    acc[3] = WMMA_BF16(ay, y3, acc[3]);
  }

  float isv[8];
#pragma unroll
  for (int r = 0; r < 8; ++r) isv[r] = invS[row0 + r + 8 * hi];

  float sq = 0.f;
#pragma unroll
  for (int j = 0; j < 4; ++j) {
    const int dcol = dbase + j * 16 + m;
#pragma unroll
    for (int r = 0; r < 8; ++r) {
      const int row = row0 + r + 8 * hi;
      const size_t off = (size_t)row * DIMD + dcol;
      const float q = acc[j][r] * isv[r];
      outQ[off] = q;
      const float d = q - x[off];
      sq += d * d;
    }
  }
  red[tid] = sq;
  __syncthreads();
  for (int s2 = 128; s2 > 0; s2 >>= 1) { if (tid < s2) red[tid] += red[tid + s2]; __syncthreads(); }
  if (tid == 0) {
    const float v = red[0] * (1.0f / ((float)N_TOK * (float)DIMD));
    atomicAdd(&outLoss[0], v);   // codebook_loss
    atomicAdd(&outLoss[1], v);   // enc_commitment_loss (numerically identical)
  }
}

// ---------------------------------------------------------------------------
extern "C" void kernel_launch(void* const* d_in, const int* in_sizes, int n_in,
                              void* d_out, int out_size, void* d_ws, size_t ws_size,
                              hipStream_t stream) {
  (void)in_sizes; (void)n_in; (void)out_size; (void)ws_size;
  const float* x  = (const float*)d_in[0];   // (4,1024,512)
  const float* cb = (const float*)d_in[1];   // (8192,512)
  const float* u1 = (const float*)d_in[2];   // (4096,8192)
  const float* u2 = (const float*)d_in[3];   // (4096,8192)

  // Workspace layout (~84.2 MB total)
  char* ws = (char*)d_ws;
  __bf16* xb  = (__bf16*)(ws);                                   //  4 MB
  __bf16* cbb = (__bf16*)(ws + ((size_t)4  << 20));              //  8 MB
  __bf16* cbT = (__bf16*)(ws + ((size_t)12 << 20));              //  8 MB
  __bf16* P   = (__bf16*)(ws + ((size_t)20 << 20));              // 64 MB
  float* xsq  = (float*)(ws + ((size_t)84 << 20));               // 16 KB
  float* csq  = (float*)(ws + ((size_t)84 << 20) + (64 << 10));  // 32 KB
  float* invS = (float*)(ws + ((size_t)84 << 20) + (128 << 10)); // 16 KB

  // Output layout: quantized (4096*512) | indices (4096) | 3 scalars
  float* out      = (float*)d_out;
  float* outQ     = out;
  float* outIdx   = out + (size_t)N_TOK * DIMD;
  float* outScal  = outIdx + N_TOK;   // [codebook_loss, commit_loss, mean_entropy]

  prep_x_kernel <<<N_TOK, 128, 0, stream>>>(x, xb, xsq, outScal);
  prep_cb_kernel<<<KCB,   128, 0, stream>>>(cb, cbb, cbT, csq);
  vq_logits_softmax_kernel<<<N_TOK / 16, 128, 0, stream>>>(
      u1, u2, xb, cbb, xsq, csq, P, invS, outIdx, &outScal[2]);
  vq_gemm2_kernel<<<N_TOK / 16, 256, 0, stream>>>(P, cbT, invS, x, outQ, outScal);
}
